// EmoGene_16277926052224
// MI455X (gfx1250) — matrix operand order repaired and checked by reading, compiled, not run on hardware
//
#include <hip/hip_runtime.h>
#include <math.h>

// ---------------- problem constants ----------------
#define E_DIM 64
#define NH 2
#define HD 32
#define SL 156
#define TL 156
#define BN 512
#define TGTD 55
#define OUTD 55

typedef __attribute__((ext_vector_type(16))) _Float16 v16h;
typedef __attribute__((ext_vector_type(8)))  _Float16 v8h;
typedef __attribute__((ext_vector_type(8)))  float    v8f;

// ---------------- WMMA helpers (CDNA5 gfx1250, wave32) ----------------
__device__ inline v8f zero8() {
  v8f z;
#pragma unroll
  for (int i = 0; i < 8; ++i) z[i] = 0.f;
  return z;
}

__device__ inline v8f wmma_f16(v16h a, v16h b, v8f c) {
  // v_wmma_f32_16x16x32_f16
  return __builtin_amdgcn_wmma_f32_16x16x32_f16(false, a, false, b, (short)0, c,
                                                false, false);
}

__device__ inline v16h cat16(v8h lo, v8h hi) {
  return __builtin_shufflevector(lo, hi, 0, 1, 2, 3, 4, 5, 6, 7, 8, 9, 10, 11,
                                 12, 13, 14, 15);
}

// A fragment (16x32 f16, MxK), ISA 7.12.2 layout: lane holds two contiguous
// 8-half runs -> two b128 loads.  Requires ldk % 8 == 0, k0 % 32 == 0.
__device__ inline v16h load_A(const _Float16* A, int row0, long ldk, int k0) {
  int l = threadIdx.x & 31;
  const _Float16* p =
      A + (long)(row0 + (l & 15)) * ldk + k0 + ((l & 16) ? 8 : 0);
  v8h lo = *(const v8h*)p;
  v8h hi = *(const v8h*)(p + 16);
  return cat16(lo, hi);
}

// B fragment (32x16, KxN) where B = X^T and X is (N x K) row-major f16:
// lane holds two 8-half runs (k 8h.., 16+8h..) of one X row -> two b128 loads.
__device__ inline v16h load_BT(const _Float16* X, long ldk, int n0, int k0) {
  int l = threadIdx.x & 31;
  const _Float16* p =
      X + (long)(n0 + (l & 15)) * ldk + k0 + ((l & 16) ? 8 : 0);
  v8h lo = *(const v8h*)p;
  v8h hi = *(const v8h*)(p + 16);
  return cat16(lo, hi);
}

// reductions across the 16 lanes of a half-wave (rows of C fragments)
__device__ inline float hmax16(float v) {
#pragma unroll
  for (int m = 8; m >= 1; m >>= 1) v = fmaxf(v, __shfl_xor(v, m, 32));
  return v;
}
__device__ inline float hsum16(float v) {
#pragma unroll
  for (int m = 8; m >= 1; m >>= 1) v += __shfl_xor(v, m, 32);
  return v;
}

#define WAIT_DS() asm volatile("s_wait_dscnt 0" ::: "memory")

// ---------------- generic fused GEMM: out = epi(A @ W16^T + bias) -----------
// A: (M x >=64) f16; W16: (64 x 64) zero-padded f16; K = 64 fixed.
// mode: 0 none, 1 row-l2norm (N must be 64), 2 abs.
// In-place (outH == A) is safe: A fragments are loaded before any store.
__global__ __launch_bounds__(128) void gemm_nt(
    const _Float16* A, long lda, const _Float16* __restrict__ W16,
    const float* __restrict__ bias, float* outF, _Float16* outH, int M, int N,
    int mode) {
  int wv = threadIdx.x >> 5;
  int row0 = (blockIdx.x * 4 + wv) * 16;
  if (row0 >= M) return;
  int l = threadIdx.x & 31, nl = l & 15, hb = (l & 16) ? 8 : 0;
  v16h a0 = load_A(A, row0, lda, 0);
  v16h a1 = load_A(A, row0, lda, 32);
  float y[4][8];
#pragma unroll
  for (int c = 0; c < 4; ++c) {
    v8f acc = zero8();
    acc = wmma_f16(a0, load_BT(W16, 64, c * 16, 0), acc);
    acc = wmma_f16(a1, load_BT(W16, 64, c * 16, 32), acc);
    int n = c * 16 + nl;
    float bb = bias[n < N ? n : 0];  // clamped index: no divergent load
    if (n >= N) bb = 0.f;
#pragma unroll
    for (int r = 0; r < 8; ++r) y[c][r] = acc[r] + bb;
  }
  if (mode == 1) {
#pragma unroll
    for (int r = 0; r < 8; ++r) {
      float p = y[0][r] * y[0][r] + y[1][r] * y[1][r] + y[2][r] * y[2][r] +
                y[3][r] * y[3][r];
      float ss = hsum16(p);
      float sc = 1.f / fmaxf(sqrtf(ss), 1e-12f);
      y[0][r] *= sc; y[1][r] *= sc; y[2][r] *= sc; y[3][r] *= sc;
    }
  } else if (mode == 2) {
#pragma unroll
    for (int c = 0; c < 4; ++c)
#pragma unroll
      for (int r = 0; r < 8; ++r) y[c][r] = fabsf(y[c][r]);
  }
  if (outF) {
#pragma unroll
    for (int c = 0; c < 4; ++c) {
      int n = c * 16 + nl;
      if (n < N) {
#pragma unroll
        for (int r = 0; r < 8; ++r)
          outF[(long)(row0 + r + hb) * N + n] = y[c][r];
      }
    }
  }
  if (outH) {
#pragma unroll
    for (int c = 0; c < 4; ++c) {
      int n = c * 16 + nl;
      if (n < N) {
#pragma unroll
        for (int r = 0; r < 8; ++r)
          outH[(long)(row0 + r + hb) * N + n] = (_Float16)y[c][r];
      }
    }
  }
}

// ---------------- scan step 1: ft/ch + q,k,v projections ----------------
__global__ __launch_bounds__(128) void step_proj(
    int s, const _Float16* __restrict__ src16, const _Float16* __restrict__ ht16,
    const float* __restrict__ ct, float* __restrict__ chF,
    _Float16* __restrict__ q16, _Float16* __restrict__ k16,
    _Float16* __restrict__ vT, const _Float16* __restrict__ Wf16,
    const float* __restrict__ Wf_b, const float* __restrict__ bf,
    const _Float16* __restrict__ eWq16, const float* __restrict__ eWq_b,
    const _Float16* __restrict__ eWk16, const float* __restrict__ eWk_b,
    const _Float16* __restrict__ eWv16, const float* __restrict__ eWv_b) {
  __shared__ __align__(16) _Float16 lds[4][16 * 64];
  int wv = threadIdx.x >> 5;
  int row0 = (blockIdx.x * 4 + wv) * 16;
  int l = threadIdx.x & 31, nl = l & 15, hb = (l & 16) ? 8 : 0;

  // cc = [xt, ht], K = 128 -> 4 A fragments
  v16h cc[4];
  cc[0] = load_A(src16 + (long)s * E_DIM, row0, (long)SL * E_DIM, 0);
  cc[1] = load_A(src16 + (long)s * E_DIM, row0, (long)SL * E_DIM, 32);
  cc[2] = load_A(ht16, row0, E_DIM, 0);
  cc[3] = load_A(ht16, row0, E_DIM, 32);

  // ft = sigmoid(cc @ Wf^T + Wf_b + bf); ch = ft * ct
#pragma unroll
  for (int c = 0; c < 4; ++c) {
    v8f acc = zero8();
#pragma unroll
    for (int kk = 0; kk < 4; ++kk)
      acc = wmma_f16(cc[kk], load_BT(Wf16, 128, c * 16, kk * 32), acc);
    int n = c * 16 + nl;
    float bb = Wf_b[n] + bf[n];
#pragma unroll
    for (int r = 0; r < 8; ++r) {
      int gr = row0 + r + hb;
      float ftv = 1.f / (1.f + __expf(-(acc[r] + bb)));
      float chv = ftv * ct[gr * E_DIM + n];
      chF[gr * E_DIM + n] = chv;
      lds[wv][(r + hb) * 64 + n] = (_Float16)chv;
    }
  }
  WAIT_DS();
  v16h chA0 = load_A(&lds[wv][0], 0, 64, 0);
  v16h chA1 = load_A(&lds[wv][0], 0, 64, 32);

  // q = ch @ eWq^T + b
#pragma unroll
  for (int c = 0; c < 4; ++c) {
    v8f acc = zero8();
    acc = wmma_f16(chA0, load_BT(eWq16, 64, c * 16, 0), acc);
    acc = wmma_f16(chA1, load_BT(eWq16, 64, c * 16, 32), acc);
    int n = c * 16 + nl;
    float bb = eWq_b[n];
#pragma unroll
    for (int r = 0; r < 8; ++r)
      q16[(row0 + r + hb) * E_DIM + n] = (_Float16)(acc[r] + bb);
  }

  // k = cc @ eWk^T + b (row-major); v = cc @ eWv^T + b (stored transposed)
#pragma unroll
  for (int c = 0; c < 4; ++c) {
    v8f ak = zero8(), av = zero8();
#pragma unroll
    for (int kk = 0; kk < 4; ++kk) {
      ak = wmma_f16(cc[kk], load_BT(eWk16, 128, c * 16, kk * 32), ak);
      av = wmma_f16(cc[kk], load_BT(eWv16, 128, c * 16, kk * 32), av);
    }
    int n = c * 16 + nl;
    float bk = eWk_b[n], bv = eWv_b[n];
#pragma unroll
    for (int r = 0; r < 8; ++r) {
      int gr = row0 + r + hb;
      k16[gr * E_DIM + n] = (_Float16)(ak[r] + bk);
      vT[(long)n * BN + gr] = (_Float16)(av[r] + bv);
    }
  }
}

// ---------------- scan step 2: flash batch-attention + state update ----------
__global__ __launch_bounds__(128) void step_attn(
    const _Float16* __restrict__ q16, const _Float16* __restrict__ k16,
    const _Float16* __restrict__ vT, const float* __restrict__ chF,
    float* __restrict__ ht, _Float16* __restrict__ ht16, float* __restrict__ ct,
    float* __restrict__ et, const _Float16* __restrict__ Wu16,
    const float* __restrict__ Wu_b) {
  __shared__ __align__(16) _Float16 lds[4][16 * 64];
  int wv = threadIdx.x >> 5;
  int row0 = (blockIdx.x * 4 + wv) * 16;
  int l = threadIdx.x & 31, nl = l & 15, hb = (l & 16) ? 8 : 0;

  v16h aq0 = load_A(q16, row0, E_DIM, 0);
  v16h aq1 = load_A(q16, row0, E_DIM, 32);
  v8f acc[4];
  float rmax[8], rsum[8];
#pragma unroll
  for (int c = 0; c < 4; ++c) acc[c] = zero8();
#pragma unroll
  for (int r = 0; r < 8; ++r) { rmax[r] = -3.0e38f; rsum[r] = 0.f; }

  for (int kc = 0; kc < 16; ++kc) {
    int key0 = kc * 32;
    if (kc + 1 < 16) __builtin_prefetch(k16 + (key0 + 32) * E_DIM, 0, 1);
    // logits (16 rows x 32 keys), scale 1/sqrt(64)
    v8f s0 = zero8(), s1 = zero8();
    s0 = wmma_f16(aq0, load_BT(k16, E_DIM, key0, 0), s0);
    s0 = wmma_f16(aq1, load_BT(k16, E_DIM, key0, 32), s0);
    s1 = wmma_f16(aq0, load_BT(k16, E_DIM, key0 + 16, 0), s1);
    s1 = wmma_f16(aq1, load_BT(k16, E_DIM, key0 + 16, 32), s1);
#pragma unroll
    for (int r = 0; r < 8; ++r) {
      float a0 = s0[r] * 0.125f, a1 = s1[r] * 0.125f;
      float tmax = hmax16(fmaxf(a0, a1));
      float nm = fmaxf(rmax[r], tmax);
      float al = __expf(rmax[r] - nm);
      rmax[r] = nm;
      float p0 = __expf(a0 - nm), p1 = __expf(a1 - nm);
      rsum[r] = rsum[r] * al + hsum16(p0 + p1);
      acc[0][r] *= al; acc[1][r] *= al; acc[2][r] *= al; acc[3][r] *= al;
      lds[wv][(r + hb) * 64 + nl] = (_Float16)p0;
      lds[wv][(r + hb) * 64 + 16 + nl] = (_Float16)p1;
    }
    WAIT_DS();
    v16h pa = load_A(&lds[wv][0], 0, 64, 0);  // 16x32 probability tile
#pragma unroll
    for (int c = 0; c < 4; ++c)
      acc[c] = wmma_f16(pa, load_BT(vT, BN, c * 16, key0), acc[c]);
  }

  float inv[8];
#pragma unroll
  for (int r = 0; r < 8; ++r) inv[r] = 1.f / rsum[r];
  // ao -> state update, stage ao for Wu GEMM
#pragma unroll
  for (int c = 0; c < 4; ++c) {
    int n = c * 16 + nl;
#pragma unroll
    for (int r = 0; r < 8; ++r) {
      float ao = acc[c][r] * inv[r];
      int gr = row0 + r + hb;
      ht[gr * E_DIM + n] = ao;
      ht16[gr * E_DIM + n] = (_Float16)ao;
      ct[gr * E_DIM + n] = chF[gr * E_DIM + n] + ao;
      lds[wv][(r + hb) * 64 + n] = (_Float16)ao;
    }
  }
  WAIT_DS();
  v16h ao0 = load_A(&lds[wv][0], 0, 64, 0);
  v16h ao1 = load_A(&lds[wv][0], 0, 64, 32);
#pragma unroll
  for (int c = 0; c < 4; ++c) {
    v8f u = zero8();
    u = wmma_f16(ao0, load_BT(Wu16, 64, c * 16, 0), u);
    u = wmma_f16(ao1, load_BT(Wu16, 64, c * 16, 32), u);
    int n = c * 16 + nl;
    float bb = Wu_b[n];
#pragma unroll
    for (int r = 0; r < 8; ++r) {
      int gr = row0 + r + hb;
      float ut = 1.f / (1.f + __expf(-(u[r] + bb)));
      et[gr * E_DIM + n] *= ut;
    }
  }
}

// ---------------- MHA: per-head 512x512 attention (t-invariant) -------------
__global__ __launch_bounds__(128) void attn_head(const _Float16* __restrict__ qh,
                                                 const _Float16* __restrict__ kh,
                                                 _Float16* __restrict__ P,
                                                 float* __restrict__ rsumb) {
  int wv = threadIdx.x >> 5;
  int w = blockIdx.x * 4 + wv;  // 64 waves: h(2) x qtile(32)
  int h = w >> 5, qt = w & 31;
  int row0 = qt * 16;
  int l = threadIdx.x & 31, nl = l & 15, hb = (l & 16) ? 8 : 0;
  const float scale = 0.17677669529663687f;  // 1/sqrt(32)
  v16h a = load_A(qh, row0, 64, h * HD);     // K = 32 (head slice)
  float rmax[8];
#pragma unroll
  for (int r = 0; r < 8; ++r) rmax[r] = -3.0e38f;
  for (int kt = 0; kt < 32; ++kt) {
    v8f sf = wmma_f16(a, load_BT(kh, 64, kt * 16, h * HD), zero8());
#pragma unroll
    for (int r = 0; r < 8; ++r) rmax[r] = fmaxf(rmax[r], sf[r] * scale);
  }
#pragma unroll
  for (int r = 0; r < 8; ++r) rmax[r] = hmax16(rmax[r]);
  float rs[8];
#pragma unroll
  for (int r = 0; r < 8; ++r) rs[r] = 0.f;
  for (int kt = 0; kt < 32; ++kt) {
    v8f sf = wmma_f16(a, load_BT(kh, 64, kt * 16, h * HD), zero8());
#pragma unroll
    for (int r = 0; r < 8; ++r) {
      float p = __expf(sf[r] * scale - rmax[r]);
      rs[r] += p;
      P[((long)h * BN + row0 + r + hb) * BN + kt * 16 + nl] = (_Float16)p;
    }
  }
#pragma unroll
  for (int r = 0; r < 8; ++r) rs[r] = hsum16(rs[r]);
  if (nl == 0) {
#pragma unroll
    for (int r = 0; r < 8; ++r) rsumb[h * BN + row0 + r + hb] = rs[r];
  }
}

// mo[b,t,h*32+d] = (P[h] @ vpT[t,:,:]^T) / rowsum
__global__ __launch_bounds__(128) void pv_kernel(const _Float16* __restrict__ P,
                                                 const _Float16* __restrict__ vpT,
                                                 const float* __restrict__ rsumb,
                                                 _Float16* __restrict__ mo) {
  int wv = threadIdx.x >> 5;
  int w = blockIdx.x * 4 + wv;  // t(156) x h(2) x qtile(32)
  int qt = w & 31, h = (w >> 5) & 1, t = w >> 6;
  int row0 = qt * 16;
  int l = threadIdx.x & 31, nl = l & 15, hb = (l & 16) ? 8 : 0;
  const _Float16* Ph = P + (long)h * BN * BN;
  const _Float16* vb = vpT + (long)t * E_DIM * BN;  // rows = col, stride BN
  v8f a0 = zero8(), a1 = zero8();
  for (int kc = 0; kc < 16; ++kc) {
    v16h pa = load_A(Ph, row0, BN, kc * 32);
    a0 = wmma_f16(pa, load_BT(vb, BN, h * HD, kc * 32), a0);
    a1 = wmma_f16(pa, load_BT(vb, BN, h * HD + 16, kc * 32), a1);
  }
#pragma unroll
  for (int r = 0; r < 8; ++r) {
    int gr = row0 + r + hb;
    float is = 1.f / rsumb[h * BN + gr];
    long o = (long)gr * TL * E_DIM + (long)t * E_DIM + h * HD;
    mo[o + nl] = (_Float16)(a0[r] * is);
    mo[o + 16 + nl] = (_Float16)(a1[r] * is);
  }
}

// ---------------- small elementwise / reduction kernels ----------------
// zero-padded f32 -> f16 convert: out is (Np x 2^klog), W is (N x K)
__global__ void cvt_pad(const float* __restrict__ W, _Float16* __restrict__ out,
                        int N, int K, long total, int klog) {
  long i = (long)blockIdx.x * blockDim.x + threadIdx.x;
  if (i >= total) return;
  long n = i >> klog;
  int k = (int)(i & ((1L << klog) - 1));
  long ns = n < N ? n : (long)(N - 1);
  int ks = k < K ? k : (K - 1);
  float w = W[ns * K + ks];  // clamped index: unconditional load
  out[i] = (n < N && k < K) ? (_Float16)w : (_Float16)0.f;
}

__global__ void gather_emb(const float* __restrict__ emb,
                           const int* __restrict__ ids,
                           _Float16* __restrict__ out, long rows) {
  long i = (long)blockIdx.x * blockDim.x + threadIdx.x;
  if (i >= rows * E_DIM) return;
  long r = i >> 6;
  int k = (int)(i & 63);
  out[i] = (_Float16)emb[(long)ids[r] * E_DIM + k];
}

__global__ void init_state(float* ht, _Float16* ht16, float* ct, float* et,
                           const float* __restrict__ et0) {
  int i = blockIdx.x * blockDim.x + threadIdx.x;
  if (i >= BN * E_DIM) return;
  ht[i] = 0.f;
  ht16[i] = (_Float16)0.f;
  ct[i] = 0.f;
  et[i] = et0[i];
}

__global__ void rownorm16(const float* __restrict__ in,
                          _Float16* __restrict__ out, int M, float ps) {
  int wv = threadIdx.x >> 5;
  int row = blockIdx.x * 4 + wv;
  if (row >= M) return;
  int l = threadIdx.x & 31;
  float v0 = in[(long)row * 64 + l] * ps;
  float v1 = in[(long)row * 64 + 32 + l] * ps;
  float ss = v0 * v0 + v1 * v1;
#pragma unroll
  for (int m = 16; m >= 1; m >>= 1) ss += __shfl_xor(ss, m, 32);
  float sc = 1.f / fmaxf(sqrtf(ss), 1e-12f);
  out[(long)row * 64 + l] = (_Float16)(v0 * sc);
  out[(long)row * 64 + 32 + l] = (_Float16)(v1 * sc);
}

__global__ void add_etpos16(_Float16* x, const _Float16* __restrict__ etn,
                            const _Float16* __restrict__ posn) {
  long i = (long)blockIdx.x * blockDim.x + threadIdx.x;
  if (i >= (long)BN * TL * E_DIM) return;
  int k = (int)(i & 63);
  long r = i >> 6;
  int t = (int)(r % TL);
  long b = r / TL;
  x[i] = (_Float16)((float)x[i] + (float)etn[b * 64 + k] +
                    (float)posn[t * 64 + k]);
}

// vp (b, t, 64) -> vpT (t, 64, 512)
__global__ void transpose_vp(const _Float16* __restrict__ vp,
                             _Float16* __restrict__ vpT) {
  long i = (long)blockIdx.x * blockDim.x + threadIdx.x;
  if (i >= (long)BN * TL * E_DIM) return;
  int col = (int)(i & 63);
  long r = i >> 6;
  int t = (int)(r % TL);
  long b = r / TL;
  vpT[((long)t * E_DIM + col) * BN + b] = vp[i];
}

__global__ void mm_init(unsigned* mm) {
  if (threadIdx.x == 0) {
    mm[0] = 0x7f800000u; mm[1] = 0u;
    mm[2] = 0x7f800000u; mm[3] = 0u;
  }
}

__global__ void mm_reduce(const float* __restrict__ x, unsigned* mm, long n,
                          int N) {
  long i0 = (long)blockIdx.x * blockDim.x + threadIdx.x;
  long stride = (long)gridDim.x * blockDim.x;
  unsigned mn1 = 0x7f800000u, mx1 = 0u, mn2 = 0x7f800000u, mx2 = 0u;
  for (long i = i0; i < n; i += stride) {
    int c = (int)(i % N);
    unsigned u = __float_as_uint(x[i]);  // all values >= 0 after abs
    bool g2 = (c == 54) || (c % 3 == 2);
    if (g2) { if (u < mn2) mn2 = u; if (u > mx2) mx2 = u; }
    else    { if (u < mn1) mn1 = u; if (u > mx1) mx1 = u; }
  }
#pragma unroll
  for (int m = 16; m >= 1; m >>= 1) {
    unsigned t;
    t = __shfl_xor(mn1, m, 32); if (t < mn1) mn1 = t;
    t = __shfl_xor(mx1, m, 32); if (t > mx1) mx1 = t;
    t = __shfl_xor(mn2, m, 32); if (t < mn2) mn2 = t;
    t = __shfl_xor(mx2, m, 32); if (t > mx2) mx2 = t;
  }
  if ((threadIdx.x & 31) == 0) {
    atomicMin(&mm[0], mn1); atomicMax(&mm[1], mx1);
    atomicMin(&mm[2], mn2); atomicMax(&mm[3], mx2);
  }
}

__global__ void final_norm(float* x, const unsigned* __restrict__ mm, long n,
                           int N) {
  long i = (long)blockIdx.x * blockDim.x + threadIdx.x;
  if (i >= n) return;
  float mn1 = __uint_as_float(mm[0]), mx1 = __uint_as_float(mm[1]);
  float mn2 = __uint_as_float(mm[2]), mx2 = __uint_as_float(mm[3]);
  int c = (int)(i % N);
  float v = x[i];
  if ((c == 54) || (c % 3 == 2)) x[i] = (v - mn2) / (mx2 - mn2);
  else x[i] = (v - mn1) / (mx1 - mn1) * 750.f + 100.f;
}

// ---------------- workspace layout (bytes) ----------------
static constexpr size_t algn(size_t x) { return (x + 255) & ~(size_t)255; }
static constexpr size_t ROWS = (size_t)BN * SL;  // 79872 (== B*T)
static constexpr size_t OFF_M16  = 0;  // src f16 during scan; mo/mo2 after
static constexpr size_t OFF_BUF1 = algn(OFF_M16 + ROWS * E_DIM * 2);   // emb16 / tgt chain f16
static constexpr size_t OFF_VP   = algn(OFF_BUF1 + ROWS * E_DIM * 2);
static constexpr size_t OFF_VPT  = algn(OFF_VP + ROWS * E_DIM * 2);
static constexpr size_t OFF_P    = algn(OFF_VPT + ROWS * E_DIM * 2);
static constexpr size_t OFF_HT   = algn(OFF_P + (size_t)NH * BN * BN * 2);
static constexpr size_t OFF_HT16 = algn(OFF_HT + BN * E_DIM * 4);
static constexpr size_t OFF_CT   = algn(OFF_HT16 + BN * E_DIM * 2);
static constexpr size_t OFF_ET   = algn(OFF_CT + BN * E_DIM * 4);
static constexpr size_t OFF_CH   = algn(OFF_ET + BN * E_DIM * 4);
static constexpr size_t OFF_Q16  = algn(OFF_CH + BN * E_DIM * 4);
static constexpr size_t OFF_K16  = algn(OFF_Q16 + BN * E_DIM * 2);
static constexpr size_t OFF_VT   = algn(OFF_K16 + BN * E_DIM * 2);
static constexpr size_t OFF_HTN  = algn(OFF_VT + BN * E_DIM * 2);
static constexpr size_t OFF_ETN  = algn(OFF_HTN + BN * E_DIM * 2);
static constexpr size_t OFF_POSN = algn(OFF_ETN + BN * E_DIM * 2);
static constexpr size_t OFF_Q0   = algn(OFF_POSN + (size_t)TL * E_DIM * 2);
static constexpr size_t OFF_K0   = algn(OFF_Q0 + BN * E_DIM * 2);
static constexpr size_t OFF_QH   = algn(OFF_K0 + BN * E_DIM * 2);
static constexpr size_t OFF_KH   = algn(OFF_QH + BN * E_DIM * 2);
static constexpr size_t OFF_RS   = algn(OFF_KH + BN * E_DIM * 2);
static constexpr size_t OFF_MM   = algn(OFF_RS + (size_t)NH * BN * 4);
static constexpr size_t OFF_W    = algn(OFF_MM + 64);
// f16 weight block sub-offsets (bytes from OFF_W)
static constexpr size_t WB_WF    = 0;                    // 64x128
static constexpr size_t WB_EWK   = WB_WF + 16384;        // 64x128
static constexpr size_t WB_EWV   = WB_EWK + 16384;       // 64x128
static constexpr size_t WB_EWQ   = WB_EWV + 16384;       // 64x64
static constexpr size_t WB_WU    = WB_EWQ + 8192;
static constexpr size_t WB_TWQ   = WB_WU + 8192;
static constexpr size_t WB_TWK   = WB_TWQ + 8192;
static constexpr size_t WB_TWV   = WB_TWK + 8192;
static constexpr size_t WB_SRCL  = WB_TWV + 8192;
static constexpr size_t WB_TGTL  = WB_SRCL + 8192;       // 64x64 (from 64x55)
static constexpr size_t WB_OUTW  = WB_TGTL + 8192;       // 64x64 (from 55x64)
static constexpr size_t WB_MHAIN = WB_OUTW + 8192;       // 192x64
static constexpr size_t WB_MHAOUT = WB_MHAIN + 24576;    // 64x64

extern "C" void kernel_launch(void* const* d_in, const int* in_sizes, int n_in,
                              void* d_out, int out_size, void* d_ws,
                              size_t ws_size, hipStream_t stream) {
  (void)in_sizes; (void)n_in; (void)out_size; (void)ws_size;
  const int*   src_ids   = (const int*)d_in[0];
  const float* tgt_input = (const float*)d_in[1];
  const float* et0       = (const float*)d_in[2];
  const float* src_emb   = (const float*)d_in[3];
  const float* pos_emb   = (const float*)d_in[4];
  const float* Wf_w = (const float*)d_in[5];
  const float* Wf_b = (const float*)d_in[6];
  const float* bf   = (const float*)d_in[7];
  const float* eWq_w = (const float*)d_in[8];  const float* eWq_b = (const float*)d_in[9];
  const float* eWk_w = (const float*)d_in[10]; const float* eWk_b = (const float*)d_in[11];
  const float* eWv_w = (const float*)d_in[12]; const float* eWv_b = (const float*)d_in[13];
  const float* Wu_w  = (const float*)d_in[14]; const float* Wu_b  = (const float*)d_in[15];
  const float* tWq_w = (const float*)d_in[16]; const float* tWq_b = (const float*)d_in[17];
  const float* tWk_w = (const float*)d_in[18]; const float* tWk_b = (const float*)d_in[19];
  const float* tWv_w = (const float*)d_in[20]; const float* tWv_b = (const float*)d_in[21];
  const float* src_lin_w = (const float*)d_in[22]; const float* src_lin_b = (const float*)d_in[23];
  const float* tgt_lin_w = (const float*)d_in[24]; const float* tgt_lin_b = (const float*)d_in[25];
  const float* out_w = (const float*)d_in[26]; const float* out_b = (const float*)d_in[27];
  const float* mha_in_w = (const float*)d_in[28]; const float* mha_in_b = (const float*)d_in[29];
  const float* mha_out_w = (const float*)d_in[30]; const float* mha_out_b = (const float*)d_in[31];

  char* ws = (char*)d_ws;
  _Float16* m16  = (_Float16*)(ws + OFF_M16);
  _Float16* buf1 = (_Float16*)(ws + OFF_BUF1);
  _Float16* vp   = (_Float16*)(ws + OFF_VP);
  _Float16* vpT  = (_Float16*)(ws + OFF_VPT);
  _Float16* P    = (_Float16*)(ws + OFF_P);
  float* ht = (float*)(ws + OFF_HT);
  _Float16* ht16 = (_Float16*)(ws + OFF_HT16);
  float* ct = (float*)(ws + OFF_CT);
  float* et = (float*)(ws + OFF_ET);
  float* chF = (float*)(ws + OFF_CH);
  _Float16* q16 = (_Float16*)(ws + OFF_Q16);
  _Float16* k16 = (_Float16*)(ws + OFF_K16);
  _Float16* vT  = (_Float16*)(ws + OFF_VT);
  _Float16* htn = (_Float16*)(ws + OFF_HTN);
  _Float16* etn = (_Float16*)(ws + OFF_ETN);
  _Float16* posn = (_Float16*)(ws + OFF_POSN);
  _Float16* Q0 = (_Float16*)(ws + OFF_Q0);
  _Float16* K0 = (_Float16*)(ws + OFF_K0);
  _Float16* qh = (_Float16*)(ws + OFF_QH);
  _Float16* kh = (_Float16*)(ws + OFF_KH);
  float* rsumb = (float*)(ws + OFF_RS);
  unsigned* mm = (unsigned*)(ws + OFF_MM);
  _Float16* wWf    = (_Float16*)(ws + OFF_W + WB_WF);
  _Float16* wEWk   = (_Float16*)(ws + OFF_W + WB_EWK);
  _Float16* wEWv   = (_Float16*)(ws + OFF_W + WB_EWV);
  _Float16* wEWq   = (_Float16*)(ws + OFF_W + WB_EWQ);
  _Float16* wWu    = (_Float16*)(ws + OFF_W + WB_WU);
  _Float16* wTWq   = (_Float16*)(ws + OFF_W + WB_TWQ);
  _Float16* wTWk   = (_Float16*)(ws + OFF_W + WB_TWK);
  _Float16* wTWv   = (_Float16*)(ws + OFF_W + WB_TWV);
  _Float16* wSrcl  = (_Float16*)(ws + OFF_W + WB_SRCL);
  _Float16* wTgtl  = (_Float16*)(ws + OFF_W + WB_TGTL);
  _Float16* wOutw  = (_Float16*)(ws + OFF_W + WB_OUTW);
  _Float16* wMhain = (_Float16*)(ws + OFF_W + WB_MHAIN);
  _Float16* wMhaout = (_Float16*)(ws + OFF_W + WB_MHAOUT);

  const long gelems = (long)ROWS * E_DIM;
  const int bigGrid = (int)((gelems + 255) / 256);
  const int gemmBig = (int)((ROWS + 63) / 64);

  // 0. pre-convert all weights to zero-padded f16 (one-time, tiny)
  cvt_pad<<<32, 256, 0, stream>>>(Wf_w,  wWf,  64, 128, 8192, 7);
  cvt_pad<<<32, 256, 0, stream>>>(eWk_w, wEWk, 64, 128, 8192, 7);
  cvt_pad<<<32, 256, 0, stream>>>(eWv_w, wEWv, 64, 128, 8192, 7);
  cvt_pad<<<16, 256, 0, stream>>>(eWq_w, wEWq, 64, 64, 4096, 6);
  cvt_pad<<<16, 256, 0, stream>>>(Wu_w,  wWu,  64, 64, 4096, 6);
  cvt_pad<<<16, 256, 0, stream>>>(tWq_w, wTWq, 64, 64, 4096, 6);
  cvt_pad<<<16, 256, 0, stream>>>(tWk_w, wTWk, 64, 64, 4096, 6);
  cvt_pad<<<16, 256, 0, stream>>>(tWv_w, wTWv, 64, 64, 4096, 6);
  cvt_pad<<<16, 256, 0, stream>>>(src_lin_w, wSrcl, 64, 64, 4096, 6);
  cvt_pad<<<16, 256, 0, stream>>>(tgt_lin_w, wTgtl, 64, TGTD, 4096, 6);
  cvt_pad<<<16, 256, 0, stream>>>(out_w, wOutw, OUTD, 64, 4096, 6);
  cvt_pad<<<48, 256, 0, stream>>>(mha_in_w, wMhain, 192, 64, 12288, 6);
  cvt_pad<<<16, 256, 0, stream>>>(mha_out_w, wMhaout, 64, 64, 4096, 6);

  // 1. src = l2n(emb[src_ids] @ src_lin^T + b)  -> f16
  gather_emb<<<bigGrid, 256, 0, stream>>>(src_emb, src_ids, buf1, (long)ROWS);
  gemm_nt<<<gemmBig, 128, 0, stream>>>(buf1, 64, wSrcl, src_lin_b, nullptr, m16,
                                       (int)ROWS, 64, 1);

  // 2. recurrent scan over SRC_LEN steps
  init_state<<<(BN * E_DIM + 255) / 256, 256, 0, stream>>>(ht, ht16, ct, et, et0);
  for (int s = 0; s < SL; ++s) {
    step_proj<<<8, 128, 0, stream>>>(s, m16, ht16, ct, chF, q16, k16, vT, wWf,
                                     Wf_b, bf, wEWq, eWq_b, wEWk, eWk_b, wEWv,
                                     eWv_b);
    step_attn<<<8, 128, 0, stream>>>(q16, k16, vT, chF, ht, ht16, ct, et, wWu,
                                     Wu_b);
  }

  // 3. normalizations + broadcast projections (all f16 intermediates)
  rownorm16<<<(BN + 3) / 4, 128, 0, stream>>>(ht, htn, BN, 1.f);
  rownorm16<<<(BN + 3) / 4, 128, 0, stream>>>(et, etn, BN, 1.f / (SL * 100.f));
  rownorm16<<<(TL + 3) / 4, 128, 0, stream>>>(pos_emb, posn, TL, 1.f);
  gemm_nt<<<8, 128, 0, stream>>>(htn, 64, wTWq, tWq_b, nullptr, Q0, BN, 64, 1);
  gemm_nt<<<8, 128, 0, stream>>>(htn, 64, wTWk, tWk_b, nullptr, K0, BN, 64, 1);
  gemm_nt<<<8, 128, 0, stream>>>(Q0, 64, wMhain, mha_in_b, nullptr, qh, BN, 64, 0);
  gemm_nt<<<8, 128, 0, stream>>>(K0, 64, wMhain + 64 * 64, mha_in_b + 64,
                                 nullptr, kh, BN, 64, 0);

  // 4. value path: tgt (padded f16) -> +et,pos -> V -> vp -> vpT
  cvt_pad<<<bigGrid, 256, 0, stream>>>(tgt_input, buf1, (int)ROWS, TGTD, gelems, 6);
  gemm_nt<<<gemmBig, 128, 0, stream>>>(buf1, 64, wTgtl, tgt_lin_b, nullptr,
                                       buf1, (int)ROWS, 64, 1);  // tgt in place
  add_etpos16<<<bigGrid, 256, 0, stream>>>(buf1, etn, posn);
  gemm_nt<<<gemmBig, 128, 0, stream>>>(buf1, 64, wTWv, tWv_b, nullptr, buf1,
                                       (int)ROWS, 64, 1);  // V in place
  gemm_nt<<<gemmBig, 128, 0, stream>>>(buf1, 64, wMhain + 128 * 64,
                                       mha_in_b + 128, nullptr, vp, (int)ROWS,
                                       64, 0);
  transpose_vp<<<bigGrid, 256, 0, stream>>>(vp, vpT);

  // 5. MHA (attention is t-invariant): P[h] once, then P @ vp per t
  attn_head<<<16, 128, 0, stream>>>(qh, kh, P, rsumb);
  pv_kernel<<<(TL * NH * 32) / 4, 128, 0, stream>>>(P, vpT, rsumb, m16);

  // 6. output projections (+abs) -> d_out, then global min/max normalize
  gemm_nt<<<gemmBig, 128, 0, stream>>>(m16, 64, wMhaout, mha_out_b, nullptr,
                                       m16, (int)ROWS, 64, 0);  // mo2 in place
  gemm_nt<<<gemmBig, 128, 0, stream>>>(m16, 64, wOutw, out_b, (float*)d_out,
                                       nullptr, (int)ROWS, OUTD, 2);
  const long n_out = (long)ROWS * OUTD;
  mm_init<<<1, 32, 0, stream>>>(mm);
  mm_reduce<<<2048, 256, 0, stream>>>((const float*)d_out, mm, n_out, OUTD);
  final_norm<<<(int)((n_out + 255) / 256), 256, 0, stream>>>((float*)d_out, mm,
                                                             n_out, OUTD);
}
